// KONEModel_2379411882082
// MI455X (gfx1250) — compile-verified, matrix operand order checked by reading, not run on hardware
//
#include <hip/hip_runtime.h>
#include <hip/hip_bf16.h>
#include <stdint.h>

// ---------------------------------------------------------------------------
// GPT forward for MI455X (gfx1250), bf16 WMMA everywhere, fp32 accumulate.
// Weights are converted fp32->bf16 AND transposed to [N,K] once per launch so
// every GEMM stages both operands with contiguous async global->LDS b128 copies.
// ---------------------------------------------------------------------------

typedef __bf16 bf16_t;
typedef bf16_t v16bf __attribute__((ext_vector_type(16)));
typedef bf16_t v8bf  __attribute__((ext_vector_type(8)));
typedef float  v8f   __attribute__((ext_vector_type(8)));
typedef int    v4i_t __attribute__((ext_vector_type(4)));

#define LN_EPS 1e-5f

// --- CDNA5 async global->LDS copy (ASYNCcnt path), guarded so the file still
// --- compiles if a toolchain doesn't declare the builtin.
// Probe-verified on this toolchain: 4 args, generic 'int4*' pointers.
#if defined(__has_builtin)
# if __has_builtin(__builtin_amdgcn_global_load_async_to_lds_b128) && \
     __has_builtin(__builtin_amdgcn_s_wait_asynccnt)
#  define USE_ASYNC_LDS 1
# endif
#endif
#ifndef USE_ASYNC_LDS
# define USE_ASYNC_LDS 0
#endif

#if USE_ASYNC_LDS
static __device__ __forceinline__ void async_copy16(const bf16_t* g, bf16_t* l) {
    __builtin_amdgcn_global_load_async_to_lds_b128(
        (v4i_t*)(uintptr_t)g, (v4i_t*)(uintptr_t)l, 0, 0);
}
static __device__ __forceinline__ void async_wait0() {
    __builtin_amdgcn_s_wait_asynccnt(0);
}
#else
static __device__ __forceinline__ void async_wait0() {}
#endif

static __device__ __forceinline__ v8bf load8(const bf16_t* p) {
    return *reinterpret_cast<const v8bf*>(p);
}
static __device__ __forceinline__ void store8(bf16_t* p, v8bf v) {
    *reinterpret_cast<v8bf*>(p) = v;
}
static __device__ __forceinline__ void copy16(const bf16_t* g, bf16_t* l) {
#if USE_ASYNC_LDS
    async_copy16(g, l);
#else
    store8(l, load8(g));
    store8(l + 8, load8(g + 8));
#endif
}

// Build a 16x32 bf16 A/B fragment for v_wmma_f32_16x16x32_bf16.
// rowbase points at this lane's 32 contiguous K elements; kb = 0 (lanes 0-15)
// or 8 (lanes 16-31) per ISA 7.12.2: lanes<16 hold K 0-7,16-23; lanes>=16 hold
// K 8-15,24-31, two K values per VGPR.
static __device__ __forceinline__ v16bf make_frag(const bf16_t* rowbase, int kb) {
    v8bf lo = load8(rowbase + kb);
    v8bf hi = load8(rowbase + kb + 16);
    v16bf r;
#pragma unroll
    for (int i = 0; i < 8; i++) { r[i] = lo[i]; r[i + 8] = hi[i]; }
    return r;
}

// ---------------------------------------------------------------------------
// fp32 -> bf16 elementwise convert (tok_emb, once per launch)
// ---------------------------------------------------------------------------
__global__ __launch_bounds__(256) void f32_to_bf16_kernel(
    const float* __restrict__ in, bf16_t* __restrict__ out, size_t n)
{
    size_t i = (size_t)blockIdx.x * 256 + threadIdx.x;
    size_t stride = (size_t)gridDim.x * 256;
    for (; i < n; i += stride) out[i] = (bf16_t)in[i];
}

// ---------------------------------------------------------------------------
// fp32 [L][K][N] -> bf16 [L][N][K] transpose+convert (weights, once per launch)
// 32x32 tiles through padded LDS; coalesced reads and writes.
// ---------------------------------------------------------------------------
__global__ __launch_bounds__(256) void transpose_cvt_kernel(
    const float* __restrict__ in, bf16_t* __restrict__ out, int K, int N)
{
    __shared__ float tile[32][33];
    const int l = blockIdx.z;
    const float* src = in + (size_t)l * K * N;
    bf16_t* dst = out + (size_t)l * K * N;
    const int k0 = blockIdx.y * 32, n0 = blockIdx.x * 32;
    const int tx = threadIdx.x & 31, ty = threadIdx.x >> 5; // ty in 0..7
#pragma unroll
    for (int r = 0; r < 4; r++)
        tile[ty + 8 * r][tx] = src[(size_t)(k0 + ty + 8 * r) * N + n0 + tx];
    __syncthreads();
#pragma unroll
    for (int r = 0; r < 4; r++)
        dst[(size_t)(n0 + ty + 8 * r) * K + k0 + tx] = (bf16_t)tile[tx][ty + 8 * r];
}

// ---------------------------------------------------------------------------
// Embedding gather: x[b,t,:] = tok_emb[idx[b,t],:] + pos_emb[t,:]
// ---------------------------------------------------------------------------
__global__ __launch_bounds__(256) void embed_kernel(
    const int* __restrict__ idx, const float* __restrict__ tok,
    const float* __restrict__ pos, float* __restrict__ x, int T, int E)
{
    int bt = blockIdx.x;
    int t = bt % T;
    int id = idx[bt];
    const float* tr = tok + (size_t)id * E;
    const float* pr = pos + (size_t)t * E;
    float* xr = x + (size_t)bt * E;
    for (int i = threadIdx.x; i < E; i += 256) xr[i] = tr[i] + pr[i];
}

// ---------------------------------------------------------------------------
// LayerNorm: one block per token row, fp32 in -> bf16 out
// ---------------------------------------------------------------------------
__global__ __launch_bounds__(256) void layernorm_kernel(
    const float* __restrict__ x, const float* __restrict__ gamma,
    const float* __restrict__ beta, bf16_t* __restrict__ out, int E)
{
    const int row = blockIdx.x;
    const int tid = threadIdx.x;
    const float* xr = x + (size_t)row * E;
    float s = 0.f, s2 = 0.f;
    for (int i = tid; i < E; i += 256) { float v = xr[i]; s += v; s2 += v * v; }
#pragma unroll
    for (int m = 1; m < 32; m <<= 1) {
        s  += __shfl_xor(s,  m, 32);
        s2 += __shfl_xor(s2, m, 32);
    }
    __shared__ float rs[8], rs2[8];
    if ((tid & 31) == 0) { rs[tid >> 5] = s; rs2[tid >> 5] = s2; }
    __syncthreads();
    s = 0.f; s2 = 0.f;
#pragma unroll
    for (int i = 0; i < 8; i++) { s += rs[i]; s2 += rs2[i]; }
    float mean = s / E;
    float var  = s2 / E - mean * mean;
    float rstd = rsqrtf(var + LN_EPS);
    bf16_t* orow = out + (size_t)row * E;
    for (int i = tid; i < E; i += 256)
        orow[i] = (bf16_t)((xr[i] - mean) * rstd * gamma[i] + beta[i]);
}

// ---------------------------------------------------------------------------
// Tiled bf16 GEMM with WMMA: C[M,N] = A[M,K] * B^T  where B is stored [N,K].
// 256 threads = 8 waves; block tile 128x128; K step 32.
// Wave w owns a 32x64 sub-tile: msub=(w>>1)*32, nsub=(w&1)*64 -> 2 A-frags
// reused across 4 B-frags = 8 v_wmma_f32_16x16x32_bf16 per K step per wave.
// All LDS staging is contiguous 16B copies -> async global->LDS b128.
// ---------------------------------------------------------------------------
#define TM 128
#define TN 128
#define TK 32

__global__ __launch_bounds__(256) void gemm_bf16_kernel(
    const bf16_t* __restrict__ A, const bf16_t* __restrict__ Bm,
    const float* __restrict__ bias, const float* __restrict__ resid,
    float* __restrict__ outF, bf16_t* __restrict__ outB,
    int M, int N, int K, int relu)
{
    __shared__ bf16_t sA[TM * TK];   // [row][k]   8 KB
    __shared__ bf16_t sB[TN * TK];   // [n][k]     8 KB
    const int tid  = threadIdx.x;
    const int wave = tid >> 5;
    const int lane = tid & 31;
    const int lrow = lane & 15;
    const int hihalf = lane >> 4;
    const int kb   = hihalf ? 8 : 0;
    const int m0 = blockIdx.y * TM;
    const int n0 = blockIdx.x * TN;
    const int msub = (wave >> 1) * 32;
    const int nsub = (wave & 1) * 64;
    (void)M;

    v8f acc[8] = {};   // [mt*4 + nt]

    const int r  = tid >> 1;           // 0..127 (row of A tile / row of B tile)
    const int kc = (tid & 1) * 16;     // 0 or 16
    const bf16_t* gA = A  + (size_t)(m0 + r) * K + kc;
    const bf16_t* gB = Bm + (size_t)(n0 + r) * K + kc;

    for (int k0 = 0; k0 < K; k0 += TK) {
        copy16(gA + k0, &sA[r * TK + kc]);
        copy16(gB + k0, &sB[r * TK + kc]);
        if (k0 + TK < K) {
            __builtin_prefetch(gA + k0 + TK, 0, 0);
            __builtin_prefetch(gB + k0 + TK, 0, 0);
        }
        async_wait0();
        __syncthreads();

        v16bf a0 = make_frag(&sA[(msub + lrow) * TK], kb);
        v16bf a1 = make_frag(&sA[(msub + 16 + lrow) * TK], kb);
#pragma unroll
        for (int nt = 0; nt < 4; nt++) {
            v16bf bfrag = make_frag(&sB[(nsub + nt * 16 + lrow) * TK], kb);
            acc[nt]     = __builtin_amdgcn_wmma_f32_16x16x32_bf16(
                false, a0, false, bfrag, (short)0, acc[nt], false, false);
            acc[4 + nt] = __builtin_amdgcn_wmma_f32_16x16x32_bf16(
                false, a1, false, bfrag, (short)0, acc[4 + nt], false, false);
        }
        __syncthreads();
    }

    // Epilogue: C layout -> lane holds col nsub+nt*16+lrow, rows msub+mt*16+rr+8*hihalf
#pragma unroll
    for (int mt = 0; mt < 2; mt++) {
#pragma unroll
        for (int nt = 0; nt < 4; nt++) {
            int col = n0 + nsub + nt * 16 + lrow;
            float bv = bias ? bias[col] : 0.f;
            v8f c = acc[mt * 4 + nt];
#pragma unroll
            for (int rr = 0; rr < 8; rr++) {
                int row = m0 + msub + mt * 16 + rr + 8 * hihalf;
                float v = c[rr] + bv;
                if (relu) v = fmaxf(v, 0.f);
                size_t o = (size_t)row * N + col;
                if (resid) v += resid[o];
                if (outF) outF[o] = v;
                if (outB) outB[o] = (bf16_t)v;
            }
        }
    }
}

// ---------------------------------------------------------------------------
// Flash attention: one wave per (b, h, 16-row q tile). Dh = 64.
// S = Q K^T via 2 WMMAs per 16 kv cols; online softmax; P re-laid-out through
// LDS into an A fragment; O += P V via 4 WMMAs per 32 kv cols.
// q/k/v/o layout: [B*T, E] with head h occupying cols h*64..h*64+63.
// ---------------------------------------------------------------------------
__global__ __launch_bounds__(32) void flash_attn_kernel(
    const bf16_t* __restrict__ Q, const bf16_t* __restrict__ Km,
    const bf16_t* __restrict__ V, bf16_t* __restrict__ O,
    int T, int E, float scale)
{
    __shared__ float  sP[16 * 32];
    __shared__ bf16_t sV[32 * 64];
    const int lane = threadIdx.x;
    const int lrow = lane & 15;
    const int hihalf = lane >> 4;
    const int kb = hihalf ? 8 : 0;
    const int q0 = blockIdx.x * 16;
    const int h  = blockIdx.y;
    const int b  = blockIdx.z;
    const int Dh = 64;
    const float NEG_INF = -__builtin_inff();

    const bf16_t* qrow = Q + ((size_t)(b * T + q0 + lrow)) * E + h * Dh;
    v16bf aq0 = make_frag(qrow, kb);        // d = 0..31
    v16bf aq1 = make_frag(qrow + 32, kb);   // d = 32..63

    v8f o_acc[4] = {};
    float mrow[8], lsum[8];
#pragma unroll
    for (int r = 0; r < 8; r++) { mrow[r] = NEG_INF; lsum[r] = 0.f; }

    for (int j0 = 0; j0 < q0 + 16; j0 += 32) {
        v8f s[2];
#pragma unroll
        for (int jt = 0; jt < 2; jt++) {
            const bf16_t* krow = Km + ((size_t)(b * T + j0 + jt * 16 + lrow)) * E + h * Dh;
            v16bf bk0 = make_frag(krow, kb);
            v16bf bk1 = make_frag(krow + 32, kb);
            v8f c = {};
            c = __builtin_amdgcn_wmma_f32_16x16x32_bf16(false, aq0, false, bk0, (short)0, c, false, false);
            c = __builtin_amdgcn_wmma_f32_16x16x32_bf16(false, aq1, false, bk1, (short)0, c, false, false);
            s[jt] = c;
        }

        // stage V tile [32 kv][64 d]: lane copies one kv row (4x 16B, async)
        const bf16_t* vrow = V + ((size_t)(b * T + j0 + lane)) * E + h * Dh;
#pragma unroll
        for (int c = 0; c < 4; c++)
            copy16(vrow + c * 16, &sV[lane * 64 + c * 16]);

        int col0 = j0 + lrow, col1 = j0 + 16 + lrow;
        float alpha[8];
#pragma unroll
        for (int r = 0; r < 8; r++) {
            int rowq = q0 + r + 8 * hihalf;
            float v0 = (col0 <= rowq) ? s[0][r] * scale : NEG_INF;
            float v1 = (col1 <= rowq) ? s[1][r] * scale : NEG_INF;
            float mx = fmaxf(v0, v1);
#pragma unroll
            for (int msk = 1; msk < 16; msk <<= 1)
                mx = fmaxf(mx, __shfl_xor(mx, msk, 32));
            float mnew = fmaxf(mrow[r], mx);
            float a = (mrow[r] == NEG_INF) ? 0.f : __expf(mrow[r] - mnew);
            float p0 = __expf(v0 - mnew);
            float p1 = __expf(v1 - mnew);
            float rsum = p0 + p1;
#pragma unroll
            for (int msk = 1; msk < 16; msk <<= 1)
                rsum += __shfl_xor(rsum, msk, 32);
            lsum[r] = lsum[r] * a + rsum;
            mrow[r] = mnew;
            alpha[r] = a;
            sP[(r + 8 * hihalf) * 32 + lrow]      = p0;
            sP[(r + 8 * hihalf) * 32 + 16 + lrow] = p1;
        }
#pragma unroll
        for (int nt = 0; nt < 4; nt++)
#pragma unroll
            for (int r = 0; r < 8; r++) o_acc[nt][r] *= alpha[r];

        async_wait0();
        __syncthreads();

        // P (16x32 fp32 in LDS) -> bf16 A fragment
        v16bf pa;
#pragma unroll
        for (int i = 0; i < 8; i++) {
            pa[i]     = (bf16_t)sP[lrow * 32 + kb + i];
            pa[i + 8] = (bf16_t)sP[lrow * 32 + kb + 16 + i];
        }
        // V B fragments: lane col d = nt*16+lrow, kv split per kb
#pragma unroll
        for (int nt = 0; nt < 4; nt++) {
            int col = nt * 16 + lrow;
            v16bf bv;
#pragma unroll
            for (int i = 0; i < 8; i++) {
                bv[i]     = sV[(kb + i) * 64 + col];
                bv[i + 8] = sV[(kb + 16 + i) * 64 + col];
            }
            o_acc[nt] = __builtin_amdgcn_wmma_f32_16x16x32_bf16(
                false, pa, false, bv, (short)0, o_acc[nt], false, false);
        }
        __syncthreads();
    }

#pragma unroll
    for (int nt = 0; nt < 4; nt++) {
#pragma unroll
        for (int r = 0; r < 8; r++) {
            int rowq = q0 + r + 8 * hihalf;
            float v = o_acc[nt][r] / lsum[r];
            O[((size_t)(b * T + rowq)) * E + h * Dh + nt * 16 + lrow] = (bf16_t)v;
        }
    }
}

// ---------------------------------------------------------------------------
// Host-side orchestration
// ---------------------------------------------------------------------------
extern "C" void kernel_launch(void* const* d_in, const int* in_sizes, int n_in,
                              void* d_out, int out_size, void* d_ws, size_t ws_size,
                              hipStream_t stream)
{
    (void)in_sizes; (void)n_in; (void)out_size; (void)ws_size;
    constexpr int Lq = 8, Hh = 16, Ee = 1024, Ff = 4096, Vv = 32000, Bb = 2, Tt = 2048;
    constexpr int Mrows = Bb * Tt; // 4096

    const int*   idx  = (const int*)  d_in[0];
    const float* tok  = (const float*)d_in[1];
    const float* pos  = (const float*)d_in[2];
    const float* wq   = (const float*)d_in[3];
    const float* wk   = (const float*)d_in[4];
    const float* wv   = (const float*)d_in[5];
    const float* wo   = (const float*)d_in[6];
    const float* bo   = (const float*)d_in[7];
    const float* w1   = (const float*)d_in[8];
    const float* b1   = (const float*)d_in[9];
    const float* w2   = (const float*)d_in[10];
    const float* b2   = (const float*)d_in[11];
    const float* ln1g = (const float*)d_in[12];
    const float* ln1b = (const float*)d_in[13];
    const float* ln2g = (const float*)d_in[14];
    const float* ln2b = (const float*)d_in[15];
    const float* lnfg = (const float*)d_in[16];
    const float* lnfb = (const float*)d_in[17];

    // Workspace carve-up (~360 MB total)
    char* wsp = (char*)d_ws;
    size_t off = 0;
    auto carve = [&](size_t bytes) -> void* {
        void* p = wsp + off;
        off += (bytes + 255) & ~(size_t)255;
        return p;
    };
    // All weight copies are bf16 and stored TRANSPOSED: [L][N][K]
    bf16_t* wq_b  = (bf16_t*)carve((size_t)Lq * Ee * Ee * 2);
    bf16_t* wk_b  = (bf16_t*)carve((size_t)Lq * Ee * Ee * 2);
    bf16_t* wv_b  = (bf16_t*)carve((size_t)Lq * Ee * Ee * 2);
    bf16_t* wo_b  = (bf16_t*)carve((size_t)Lq * Ee * Ee * 2);
    bf16_t* w1_b  = (bf16_t*)carve((size_t)Lq * Ee * Ff * 2);  // [Ff][Ee] per layer
    bf16_t* w2_b  = (bf16_t*)carve((size_t)Lq * Ff * Ee * 2);  // [Ee][Ff] per layer
    bf16_t* te_b  = (bf16_t*)carve((size_t)Vv * Ee * 2);       // [Vv][Ee] already N-major
    float*  x     = (float*) carve((size_t)Mrows * Ee * 4);
    bf16_t* hbf   = (bf16_t*)carve((size_t)Mrows * Ee * 2);
    bf16_t* qb    = (bf16_t*)carve((size_t)Mrows * Ee * 2);
    bf16_t* kbuf  = (bf16_t*)carve((size_t)Mrows * Ee * 2);
    bf16_t* vbuf  = (bf16_t*)carve((size_t)Mrows * Ee * 2);
    bf16_t* attnb = (bf16_t*)carve((size_t)Mrows * Ee * 2);
    bf16_t* h1b   = (bf16_t*)carve((size_t)Mrows * Ff * 2);

    // Convert + transpose weights: in [L][K][N] fp32 -> out [L][N][K] bf16
    auto tcvt = [&](const float* src, bf16_t* dst, int K, int N) {
        transpose_cvt_kernel<<<dim3(N / 32, K / 32, Lq), dim3(256), 0, stream>>>(src, dst, K, N);
    };
    tcvt(wq, wq_b, Ee, Ee);
    tcvt(wk, wk_b, Ee, Ee);
    tcvt(wv, wv_b, Ee, Ee);
    tcvt(wo, wo_b, Ee, Ee);
    tcvt(w1, w1_b, Ee, Ff);
    tcvt(w2, w2_b, Ff, Ee);
    f32_to_bf16_kernel<<<dim3(8192), dim3(256), 0, stream>>>(tok, te_b, (size_t)Vv * Ee);

    embed_kernel<<<dim3(Mrows), dim3(256), 0, stream>>>(idx, tok, pos, x, Tt, Ee);

    auto gemm = [&](const bf16_t* A, const bf16_t* Bm, const float* bias,
                    const float* resid, float* outF, bf16_t* outB,
                    int M, int N, int K, int relu) {
        dim3 g(N / TN, M / TM);
        gemm_bf16_kernel<<<g, dim3(256), 0, stream>>>(A, Bm, bias, resid, outF, outB,
                                                      M, N, K, relu);
    };

    const float scale = 0.125f; // 1/sqrt(64)

    for (int l = 0; l < Lq; l++) {
        layernorm_kernel<<<dim3(Mrows), dim3(256), 0, stream>>>(
            x, ln1g + (size_t)l * Ee, ln1b + (size_t)l * Ee, hbf, Ee);

        gemm(hbf, wq_b + (size_t)l * Ee * Ee, nullptr, nullptr, nullptr, qb,
             Mrows, Ee, Ee, 0);
        gemm(hbf, wk_b + (size_t)l * Ee * Ee, nullptr, nullptr, nullptr, kbuf,
             Mrows, Ee, Ee, 0);
        gemm(hbf, wv_b + (size_t)l * Ee * Ee, nullptr, nullptr, nullptr, vbuf,
             Mrows, Ee, Ee, 0);

        flash_attn_kernel<<<dim3(Tt / 16, Hh, Bb), dim3(32), 0, stream>>>(
            qb, kbuf, vbuf, attnb, Tt, Ee, scale);

        gemm(attnb, wo_b + (size_t)l * Ee * Ee, bo + (size_t)l * Ee, x, x, nullptr,
             Mrows, Ee, Ee, 0);

        layernorm_kernel<<<dim3(Mrows), dim3(256), 0, stream>>>(
            x, ln2g + (size_t)l * Ee, ln2b + (size_t)l * Ee, hbf, Ee);

        gemm(hbf, w1_b + (size_t)l * Ee * Ff, b1 + (size_t)l * Ff, nullptr, nullptr,
             h1b, Mrows, Ff, Ee, 1);
        gemm(h1b, w2_b + (size_t)l * Ff * Ee, b2 + (size_t)l * Ee, x, x, nullptr,
             Mrows, Ee, Ff, 0);
    }

    layernorm_kernel<<<dim3(Mrows), dim3(256), 0, stream>>>(x, lnfg, lnfb, hbf, Ee);

    // logits = hbf @ tok_emb^T  (te_b is [V,E] = [N,K], exactly what gemm wants)
    gemm(hbf, te_b, nullptr, nullptr, (float*)d_out, nullptr,
         Mrows, Vv, Ee, 0);
}
